// Model_24292335026626
// MI455X (gfx1250) — matrix operand order repaired and checked by reading, compiled, not run on hardware
//
#include <hip/hip_runtime.h>
#include <hip/hip_bf16.h>
#include <math.h>

// ---------------------------------------------------------------------------
// S4D sequence model forward for MI455X (gfx1250, wave32, WMMA + TDM).
// Causal length-L conv as block-Toeplitz GEMM on v_wmma_f32_16x16x32_bf16.
// ---------------------------------------------------------------------------

#define BB   32
#define LL   1024
#define DD   256
#define NN   64
#define NLAY 8
#define NIT  8
#define USTRIDE 1032    // row stride in bf16 elems: 2064 B = 129*16 (b128-aligned),
                        // 516 dwords -> rows land on distinct LDS banks (4 apart)

typedef __attribute__((ext_vector_type(16))) __bf16        bf16x16;
typedef __attribute__((ext_vector_type(8)))  float         f32x8;
typedef __attribute__((ext_vector_type(4)))  unsigned int  u32x4v;
typedef __attribute__((ext_vector_type(8)))  int           i32x8v;
typedef __attribute__((ext_vector_type(4)))  int           i32x4v;

union FragU { u32x4v q[2]; bf16x16 v; };   // 32 B: one WMMA 16-bit operand per lane

__device__ __forceinline__ unsigned short f2bf(float f) {
    union { float f; unsigned int u; } c; c.f = f;
    unsigned int u = c.u;
    u += 0x7FFFu + ((u >> 16) & 1u);       // round-to-nearest-even
    return (unsigned short)(u >> 16);
}
__device__ __forceinline__ float bf2f(unsigned short h) {
    union { unsigned int u; float f; } c; c.u = ((unsigned int)h) << 16;
    return c.f;
}
__device__ __forceinline__ float gelu_exact(float x) {
    return 0.5f * x * (1.0f + erff(x * 0.70710678118654752440f));
}

// ---------------------------------------------------------------------------
// 1) Per-layer S4D conv kernels K[layer][d][l] (ZOH-discretized Vandermonde).
// ---------------------------------------------------------------------------
__global__ void k_ssm_kernel(const float* __restrict__ log_dt,
                             const float* __restrict__ log_A_real,
                             const float* __restrict__ A_imag,
                             const float* __restrict__ C_re,
                             const float* __restrict__ C_im,
                             float* __restrict__ Ks) {
    const int layer = blockIdx.x >> 8;
    const int d     = blockIdx.x & 255;
    const int tid   = threadIdx.x;
    __shared__ float sAr[NN], sAi[NN], sCr[NN], sCi[NN];
    if (tid < NN) {
        const int n = tid;
        const size_t base = ((size_t)layer * DD + d) * NN + n;
        float dt = expf(log_dt[layer * DD + d]);
        float Ar = -expf(log_A_real[base]);
        float Ai = A_imag[base];
        float ar = dt * Ar, ai = dt * Ai;       // dtA
        float mg = expf(ar);
        float s, c; sincosf(ai, &s, &c);
        float er = mg * c - 1.0f, ei = mg * s;  // exp(dtA)-1
        float inv = 1.0f / (Ar * Ar + Ai * Ai); // /A (complex)
        float qr = (er * Ar + ei * Ai) * inv;
        float qi = (ei * Ar - er * Ai) * inv;
        float Cr = C_re[base], Ci = C_im[base];
        sCr[n] = Cr * qr - Ci * qi;             // Ctilde
        sCi[n] = Cr * qi + Ci * qr;
        sAr[n] = ar; sAi[n] = ai;
    }
    __syncthreads();
    for (int l = tid; l < LL; l += 256) {
        float acc = 0.0f;
        float fl = (float)l;
        #pragma unroll 4
        for (int n = 0; n < NN; ++n) {
            float mg = expf(sAr[n] * fl);
            float s, c; sincosf(sAi[n] * fl, &s, &c);
            acc += sCr[n] * mg * c - sCi[n] * mg * s;
        }
        Ks[((size_t)layer * DD + d) * LL + l] = 2.0f * acc;
    }
}

// ---------------------------------------------------------------------------
// 2) Embedding gather: xbuf[b,l,d] = emb[x[b,l], d]
// ---------------------------------------------------------------------------
__global__ void k_embed(const int* __restrict__ x, const float* __restrict__ emb,
                        float* __restrict__ xbuf) {
    const size_t total = (size_t)BB * LL * DD;
    for (size_t i = (size_t)blockIdx.x * blockDim.x + threadIdx.x; i < total;
         i += (size_t)gridDim.x * blockDim.x) {
        size_t bl = i / DD; int d = (int)(i % DD);
        xbuf[i] = emb[(size_t)x[bl] * DD + d];
    }
}

// ---------------------------------------------------------------------------
// 3) LayerNorm over D, writing transposed bf16 [D][B][L] for the conv kernel.
// ---------------------------------------------------------------------------
__global__ void k_ln(const float* __restrict__ xin, unsigned short* __restrict__ hlnT,
                     const float* __restrict__ scale, const float* __restrict__ bias) {
    const int bl = blockIdx.x;            // b*L + l
    const int d  = threadIdx.x;
    __shared__ float red[DD];
    float v = xin[(size_t)bl * DD + d];
    red[d] = v; __syncthreads();
    for (int s = 128; s > 0; s >>= 1) { if (d < s) red[d] += red[d + s]; __syncthreads(); }
    float mu = red[0] * (1.0f / DD);
    __syncthreads();
    float dv = v - mu;
    red[d] = dv * dv; __syncthreads();
    for (int s = 128; s > 0; s >>= 1) { if (d < s) red[d] += red[d + s]; __syncthreads(); }
    float var = red[0] * (1.0f / DD);
    float o = dv * rsqrtf(var + 1e-5f) * scale[d] + bias[d];
    hlnT[(size_t)d * (BB * LL) + bl] = f2bf(o);
}

// ---------------------------------------------------------------------------
// 4) Fused block-Toeplitz causal conv + D-skip + GELU + residual (WMMA core).
//    grid = (DD, BB/16), 128 threads (4 waves). Wave w handles output chunks
//    i = w, w+4, ...; each chunk reduces over 32-wide m-chunk pairs with a
//    two-stage software pipeline of b128 fragment loads (no rotation movs).
// ---------------------------------------------------------------------------
#define LOAD_FRAG(FA, FB, P)                                     \
    do {                                                         \
        const int _pb = 4 * (P);                                 \
        (FA).q[0] = urow4[_pb + 2 * hi];                         \
        (FA).q[1] = urow4[_pb + 2 * hi + 1];                     \
        const int _c = i - 2 * (P);                              \
        (FB).q[0] = bfr4[(_c * 32 + lane) * 2 + 0];              \
        (FB).q[1] = bfr4[(_c * 32 + lane) * 2 + 1];              \
    } while (0)

__global__ void __launch_bounds__(128)
k_conv(const unsigned short* __restrict__ hlnT,  // [D][B][L] bf16 (LN output)
       const float* __restrict__ xin,            // [B][L][D] residual base
       float* __restrict__ xout,                 // [B][L][D]
       const float* __restrict__ Kc,             // [D][L] this layer's kernel
       const float* __restrict__ Dskip) {        // [D]
    const int d    = blockIdx.x;
    const int b0   = blockIdx.y * 16;
    const int tid  = threadIdx.x;
    // wave id is wave-uniform: force it scalar so loop control stays in SALU
    const int wave = __builtin_amdgcn_readfirstlane(tid >> 5);
    const int lane = tid & 31;

    extern __shared__ char smem[];
    unsigned short* u_bf = (unsigned short*)smem;                              // [16][USTRIDE]
    float*          kc   = (float*)(smem + 16 * USTRIDE * 2);                  // [1024]
    unsigned int*   bfrg = (unsigned int*)(smem + 16 * USTRIDE * 2 + LL * 4);  // [64][32][8]

    __builtin_prefetch(&hlnT[(size_t)d * (BB * LL)], 0, 0);   // global_prefetch_b8

    // --- Stage this channel's 1024 conv taps into LDS via the Tensor Data Mover.
#if __has_builtin(__builtin_amdgcn_tensor_load_to_lds)
    if (wave == 0) {
        unsigned long long ga = (unsigned long long)(const void*)(Kc + (size_t)d * LL);
        unsigned int lds_off  = (unsigned int)(unsigned long long)(void*)kc;
        // D# group0: count=1 | lds_addr | global_addr[56:0] | type=2
        u32x4v g0 = { 1u, lds_off, (unsigned int)(ga & 0xFFFFFFFFu),
                      (unsigned int)((ga >> 32) & 0x01FFFFFFu) | 0x80000000u };
        // D# group1: data_size=4B(code 2); tensor_dim0=1024; tensor_dim1=1;
        //            tile_dim0=1024; tile_dim1=1; dim0_stride=1024; dim1_stride=1024
        i32x8v g1 = { 0x00020000, 0x04000000, 0x00010000, 0x04000000,
                      0x00000001, 0x00000400, 0x04000000, 0x00000000 };
        i32x4v z4 = { 0, 0, 0, 0 };
        i32x8v z8 = { 0, 0, 0, 0, 0, 0, 0, 0 };
        __builtin_amdgcn_tensor_load_to_lds(g0, g1, z4, z4, z8, 0);
        __builtin_amdgcn_s_wait_tensorcnt(0);
    }
#else
    for (int i = tid; i < LL; i += blockDim.x) kc[i] = Kc[(size_t)d * LL + i];
#endif

    // --- 16 batch rows of bf16 LN output, b128 global->LDS (coalesced).
    {
        const unsigned short* src = hlnT + (size_t)d * (BB * LL) + (size_t)b0 * LL;
        for (int ch = tid; ch < 16 * (LL / 8); ch += blockDim.x) {  // 16B chunks
            int r = ch >> 7, c8 = ch & 127;
            u32x4v vv = *(const u32x4v*)(src + (size_t)r * LL + c8 * 8);
            *(u32x4v*)(u_bf + r * USTRIDE + c8 * 8) = vv;
        }
    }
    __syncthreads();

    // --- Pre-build the 64 Toeplitz B-fragments (32m x 16l, bf16 WMMA layout):
    // B[mq, n] = K[16*c + n - mq], zero if negative (causality).
    for (int c = wave; c < 64; c += 4) {
        const int n  = lane & 15;
        const int kq = (lane < 16) ? 0 : 16;
        unsigned int* dst = &bfrg[((size_t)c * 32 + lane) * 8];
        #pragma unroll
        for (int q = 0; q < 8; ++q) {
            int i0 = 16 * c + n - (kq + 2 * q);     // element e = 2q
            int i1 = i0 - 1;                        // element e = 2q+1
            unsigned short lo = (i0 >= 0) ? f2bf(kc[i0]) : (unsigned short)0;
            unsigned short hi = (i1 >= 0) ? f2bf(kc[i1]) : (unsigned short)0;
            dst[q] = (unsigned int)lo | ((unsigned int)hi << 16);
        }
    }
    __syncthreads();

    const float dp  = Dskip[d];
    const int   row = lane & 15;
    const int   hi  = (lane < 16) ? 0 : 1;   // selects K-halves: half hi holds k = 16*hi + s (contiguous order)
    const u32x4v* urow4 = (const u32x4v*)(u_bf + row * USTRIDE);
    const u32x4v* bfr4  = (const u32x4v*)bfrg;

    for (int i = wave; i < 64; i += 4) {
        const int nw = (i >> 1) + 1;             // number of K=32 WMMA steps
        f32x8 acc = {0.f, 0.f, 0.f, 0.f, 0.f, 0.f, 0.f, 0.f};

        // Two-stage pipeline over fragment register sets X and Y:
        //   load Y(t+1); wmma X(t); load X(t+2); wmma Y(t+1)
        FragU ax, bx, ay, by;
        LOAD_FRAG(ax, bx, 0);
        int t = 0;
        for (; t + 2 <= nw; t += 2) {
            LOAD_FRAG(ay, by, t + 1);
            acc = __builtin_amdgcn_wmma_f32_16x16x32_bf16(
                false, ax.v, false, bx.v, (short)0, acc, false, false);
            if (t + 2 < nw) LOAD_FRAG(ax, bx, t + 2);
            acc = __builtin_amdgcn_wmma_f32_16x16x32_bf16(
                false, ay.v, false, by.v, (short)0, acc, false, false);
        }
        if (nw & 1) {   // odd count: last step left in set X
            acc = __builtin_amdgcn_wmma_f32_16x16x32_bf16(
                false, ax.v, false, bx.v, (short)0, acc, false, false);
        }

        // Epilogue: y = conv + h*Dskip; out = xin + gelu(y).
        const int l = 16 * i + (lane & 15);
        #pragma unroll
        for (int q = 0; q < 8; ++q) {
            int blr = (lane < 16) ? q : (8 + q);
            float uval = bf2f(u_bf[blr * USTRIDE + l]);
            float y = acc[q] + uval * dp;
            size_t idx = ((size_t)(b0 + blr) * LL + l) * DD + d;
            xout[idx] = xin[idx] + gelu_exact(y);
        }
    }
}

// ---------------------------------------------------------------------------
// 5) t = z + h (DEQ pre-add)
// ---------------------------------------------------------------------------
__global__ void k_add(const float* __restrict__ a, const float* __restrict__ b,
                      float* __restrict__ o) {
    const size_t total = (size_t)BB * LL * DD;
    for (size_t i = (size_t)blockIdx.x * blockDim.x + threadIdx.x; i < total;
         i += (size_t)gridDim.x * blockDim.x)
        o[i] = a[i] + b[i];
}

// ---------------------------------------------------------------------------
// 6) Residual norms: acc2[0]=sum((z-prev)^2), acc2[1]=sum(z^2)
// ---------------------------------------------------------------------------
__global__ void k_norms(const float* __restrict__ z, const float* __restrict__ prev,
                        float* __restrict__ acc2) {
    const size_t total = (size_t)BB * LL * DD;
    float a = 0.f, b = 0.f;
    for (size_t i = (size_t)blockIdx.x * blockDim.x + threadIdx.x; i < total;
         i += (size_t)gridDim.x * blockDim.x) {
        float zv = z[i]; float dv = zv - prev[i];
        a += dv * dv; b += zv * zv;
    }
    __shared__ float ra[256], rb[256];
    const int t = threadIdx.x;
    ra[t] = a; rb[t] = b; __syncthreads();
    for (int s = 128; s > 0; s >>= 1) {
        if (t < s) { ra[t] += ra[t + s]; rb[t] += rb[t + s]; }
        __syncthreads();
    }
    if (t == 0) { atomicAdd(&acc2[0], ra[0]); atomicAdd(&acc2[1], rb[0]); }
}

// ---------------------------------------------------------------------------
// 7) Head: y7 = h + gelu(z); masked mean over L; logits = mean @ fc_w + fc_b.
// ---------------------------------------------------------------------------
__global__ void k_head(const int* __restrict__ x, const float* __restrict__ h,
                       const float* __restrict__ z, const float* __restrict__ fc_w,
                       const float* __restrict__ fc_b, const float* __restrict__ acc2,
                       float* __restrict__ out) {
    const int b = blockIdx.x, d = threadIdx.x;
    __shared__ float meanv[DD];
    float sum = 0.f, len = 0.f;
    for (int l = 0; l < LL; ++l) {
        float mk = (x[b * LL + l] != 0) ? 1.0f : 0.0f;
        size_t idx = ((size_t)b * LL + l) * DD + d;
        float y = h[idx] + gelu_exact(z[idx]);
        sum += y * mk; len += mk;
    }
    meanv[d] = sum / (len + 1e-8f);
    __syncthreads();
    if (d < 10) {
        float acc = fc_b[d];
        for (int k = 0; k < DD; ++k) acc += meanv[k] * fc_w[k * 10 + d];
        out[b * 10 + d] = acc;
    }
    if (b == 0 && d == 0) {
        float res = sqrtf(acc2[0]) / (sqrtf(acc2[1]) + 1e-8f);
        out[BB * 10 + 0] = res;          // loss1
        out[BB * 10 + 1] = (float)NIT;   // n_steps1
        out[BB * 10 + 2] = res;          // zhanbi1
    }
}

// ---------------------------------------------------------------------------
extern "C" void kernel_launch(void* const* d_in, const int* in_sizes, int n_in,
                              void* d_out, int out_size, void* d_ws, size_t ws_size,
                              hipStream_t stream) {
    const int*   x          = (const int*)d_in[0];
    const float* emb        = (const float*)d_in[1];
    const float* ln_scale   = (const float*)d_in[2];
    const float* ln_bias    = (const float*)d_in[3];
    const float* log_dt     = (const float*)d_in[4];
    const float* log_A_real = (const float*)d_in[5];
    const float* A_imag     = (const float*)d_in[6];
    const float* C_re       = (const float*)d_in[7];
    const float* C_im       = (const float*)d_in[8];
    const float* D_skip     = (const float*)d_in[9];
    const float* fc_w       = (const float*)d_in[10];
    const float* fc_b       = (const float*)d_in[11];
    float* out = (float*)d_out;

    // Workspace carve-up (~160 MB).
    const size_t SZ_K = (size_t)NLAY * DD * LL * sizeof(float);        // 8 MB
    const size_t SZ_T = (size_t)BB * LL * DD * sizeof(float);          // 33.5 MB
    const size_t SZ_H = (size_t)BB * LL * DD * sizeof(unsigned short); // 16.8 MB
    char* ws = (char*)d_ws;
    float*          Ks    = (float*)ws;
    float*          xbuf  = (float*)(ws + SZ_K);
    float*          za    = (float*)(ws + SZ_K + SZ_T);
    float*          zb    = (float*)(ws + SZ_K + 2 * SZ_T);
    float*          zprev = (float*)(ws + SZ_K + 3 * SZ_T);
    unsigned short* hlnT  = (unsigned short*)(ws + SZ_K + 4 * SZ_T);
    float*          acc2  = (float*)(ws + SZ_K + 4 * SZ_T + SZ_H);

    const unsigned int SMEM = 16 * USTRIDE * 2 + LL * 4 + 64 * 32 * 8 * 4; // 102,656 B

    // 1) SSM conv kernels (all 8 layers, once).
    k_ssm_kernel<<<NLAY * DD, 256, 0, stream>>>(log_dt, log_A_real, A_imag, C_re, C_im, Ks);
    // 2) Embedding.
    k_embed<<<4096, 256, 0, stream>>>(x, emb, xbuf);
    // 3) Layers 0..5.
    for (int i = 0; i < 6; ++i) {
        k_ln<<<BB * LL, DD, 0, stream>>>(xbuf, hlnT, ln_scale + i * DD, ln_bias + i * DD);
        k_conv<<<dim3(DD, BB / 16), 128, SMEM, stream>>>(hlnT, xbuf, xbuf,
                                                         Ks + (size_t)i * DD * LL,
                                                         D_skip + i * DD);
    }
    // 4) DEQ: z <- f(z + h), 8 iterations, ping-pong buffers.
    (void)hipMemsetAsync(za, 0, SZ_T, stream);
    float* zcur = za; float* znx = zb;
    for (int it = 0; it < NIT; ++it) {
        if (it == NIT - 1)
            (void)hipMemcpyAsync(zprev, zcur, SZ_T, hipMemcpyDeviceToDevice, stream);
        k_add<<<8192, 256, 0, stream>>>(zcur, xbuf, znx);
        for (int i = 6; i < 8; ++i) {
            k_ln<<<BB * LL, DD, 0, stream>>>(znx, hlnT, ln_scale + i * DD, ln_bias + i * DD);
            k_conv<<<dim3(DD, BB / 16), 128, SMEM, stream>>>(hlnT, znx, znx,
                                                             Ks + (size_t)i * DD * LL,
                                                             D_skip + i * DD);
        }
        float* t = zcur; zcur = znx; znx = t;
    }
    // 5) Residual norms + head.
    (void)hipMemsetAsync(acc2, 0, 2 * sizeof(float), stream);
    k_norms<<<2048, 256, 0, stream>>>(zcur, zprev, acc2);
    k_head<<<BB, DD, 0, stream>>>(x, xbuf, zcur, fc_w, fc_b, acc2, out);
}
